// MultiheadAttention_43585328120536
// MI455X (gfx1250) — matrix operand order repaired and checked by reading, not compiled, not run on hardware
//
#include <hip/hip_runtime.h>

// ---------------------------------------------------------------------------
// Fused causal MHA + LayerNorm + output projection for MI455X (gfx1250).
// GEMMs: v_wmma_f32_16x16x32_bf16 with double-buffered async-LDS weight tiles
// (global_load_async_to_lds_b128 + s_wait_asynccnt). Attention: flash-style
// online softmax, WMMA for QK^T and P*V, LDS relayout of P (C-tile -> A-frag).
// Workspace (~64 MB): xb, wqb/wkb/wvb/wob, qb, kb(k*1/sqrt(D)), vtb[B,H,D,S],
// of (fp32 pre-LN), lnb.
// ---------------------------------------------------------------------------

typedef __attribute__((ext_vector_type(16))) __bf16 v16bf;
typedef __attribute__((ext_vector_type(8)))  __bf16 v8bf;
typedef __attribute__((ext_vector_type(4)))  __bf16 v4bf;
typedef __attribute__((ext_vector_type(8)))  float  v8f;

union Frag {
  v16bf v;
  v8bf  h8[2];
};

static __device__ inline __bf16 f2bf(float f) {
  unsigned u = __builtin_bit_cast(unsigned, f);
  unsigned r = u + 0x7FFFu + ((u >> 16) & 1u);   // round-to-nearest-even
  unsigned short h = (unsigned short)(r >> 16);
  return __builtin_bit_cast(__bf16, h);
}

// gfx1250 async global->LDS copy (ASYNCcnt-tracked). vdst = LDS byte address
// (low 32 bits of the generic pointer), vaddr = 64-bit global address.
static __device__ inline void async_ld_b128(const void* gptr, void* ldsptr) {
  unsigned long long ga = (unsigned long long)gptr;
  unsigned lo = (unsigned)(unsigned long long)ldsptr;
  asm volatile("global_load_async_to_lds_b128 %0, %1, off"
               :: "v"(lo), "v"(ga) : "memory");
}

// --------------------------- fp32 -> bf16 convert ---------------------------
__global__ __launch_bounds__(256) void cvt_bf16(const float* __restrict__ in,
                                                __bf16* __restrict__ out, int n4) {
  int i = blockIdx.x * blockDim.x + threadIdx.x;
  if (i < n4) {
    float4 v = ((const float4*)in)[i];
    v4bf o;
    o.x = f2bf(v.x); o.y = f2bf(v.y); o.z = f2bf(v.z); o.w = f2bf(v.w);
    ((v4bf*)out)[i] = o;
  }
}

// --------------------------- WMMA GEMM: C = A @ W^T -------------------------
// A: [M,K] bf16 row-major.  W: [N,K] bf16 row-major (torch [out,in]).
// Block = 8 waves, block tile 128x128. W tile (128 rows x 32 k) is staged in
// LDS via async b128 copies, double buffered; each wave owns 16 rows of A and
// all 8 N-tiles (8 WMMAs per K-step).
// MODE 0: fp32 out row-major.  MODE 1: bf16 out, scaled.  MODE 2: Vt scatter.
#define WROWP 40  // LDS row pitch in halves (80B) to break bank conflicts

template <int MODE>
__global__ __launch_bounds__(256) void gemm_xwT(const __bf16* __restrict__ A,
                                                const __bf16* __restrict__ W,
                                                void* __restrict__ Cout,
                                                int M, int N, int K, float scale) {
  __shared__ __bf16 wtile[2][128 * WROWP];

  const int wv = threadIdx.x >> 5;
  const int lane = threadIdx.x & 31;
  const int hf = lane >> 4;
  const int li = lane & 15;

  const int nBlk = N >> 7;                    // N/128
  const int bm = blockIdx.x / nBlk;
  const int bn = blockIdx.x % nBlk;
  const int m0 = bm * 128 + wv * 16;
  const int n0 = bn * 128;

  // per-thread async-copy slice: 2 x b128 per K-step (128 rows x 64B = 8KB)
  const int crow = threadIdx.x >> 1;          // 0..127
  const int cchk = (threadIdx.x & 1) * 16;    // halves within row
  const __bf16* wsrc = W + (size_t)(n0 + crow) * K + cchk;
  __bf16* wdst0 = &wtile[0][crow * WROWP + cchk];
  __bf16* wdst1 = &wtile[1][crow * WROWP + cchk];

  v8f acc[8];
#pragma unroll
  for (int i = 0; i < 8; ++i) acc[i] = v8f{};

  const __bf16* arow = A + (size_t)(m0 + li) * K;

  // prologue: fill buffer 0
  async_ld_b128(wsrc, wdst0);
  async_ld_b128(wsrc + 8, wdst0 + 8);

  const int NK = K >> 5;
  for (int it = 0; it < NK; ++it) {
    const int k0 = it * 32;
    const int buf = it & 1;
    if (it + 1 < NK) {  // prefetch next K-slab into the other buffer
      const __bf16* src = wsrc + (k0 + 32);
      __bf16* dst = buf ? wdst0 : wdst1;
      async_ld_b128(src, dst);
      async_ld_b128(src + 8, dst + 8);
      asm volatile("s_wait_asynccnt 0x2" ::: "memory");  // current slab done
    } else {
      asm volatile("s_wait_asynccnt 0x0" ::: "memory");
    }
    __syncthreads();  // all waves' slices of wtile[buf] are resident

    Frag a;  // A layout: row=li, K in {hf*16..+16}
    a.h8[0] = *(const v8bf*)(arow + k0 + hf * 16);
    a.h8[1] = *(const v8bf*)(arow + k0 + hf * 16 + 8);
#pragma unroll
    for (int nt = 0; nt < 8; ++nt) {
      const __bf16* wl = &wtile[buf][(nt * 16 + li) * WROWP + hf * 16];
      Frag b;  // B layout: col=li, K in {hf*16..+16}
      b.h8[0] = *(const v8bf*)(wl);
      b.h8[1] = *(const v8bf*)(wl + 8);
      acc[nt] = __builtin_amdgcn_wmma_f32_16x16x32_bf16(
          false, a.v, false, b.v, (short)0, acc[nt], false, false);
    }
    __syncthreads();  // readers done before next iter overwrites this buffer
  }

  // C layout: VGPR r holds (row hf*8+r, col li) per half-wave.
  if (MODE == 0) {
    float* C = (float*)Cout;
#pragma unroll
    for (int nt = 0; nt < 8; ++nt)
#pragma unroll
      for (int r = 0; r < 8; ++r)
        C[(size_t)(m0 + hf * 8 + r) * N + n0 + nt * 16 + li] = acc[nt][r];
  } else if (MODE == 1) {
    __bf16* C = (__bf16*)Cout;
#pragma unroll
    for (int nt = 0; nt < 8; ++nt)
#pragma unroll
      for (int r = 0; r < 8; ++r)
        C[(size_t)(m0 + hf * 8 + r) * N + n0 + nt * 16 + li] =
            f2bf(acc[nt][r] * scale);
  } else {  // MODE 2: scatter V into Vt[b][h][d][s], e = d*16 + h, S=1024
    __bf16* Vt = (__bf16*)Cout;
#pragma unroll
    for (int nt = 0; nt < 8; ++nt)
#pragma unroll
      for (int r = 0; r < 8; ++r) {
        int row = m0 + hf * 8 + r;
        int bb = row >> 10, s = row & 1023;
        int e = n0 + nt * 16 + li;
        int hh = e & 15, d = e >> 4;
        Vt[(((size_t)bb * 16 + hh) * 64 + d) * 1024 + s] = f2bf(acc[nt][r]);
      }
  }
}

// --------------------------- flash attention (causal) -----------------------
// One wave per (b, h, 16-query block). Online softmax over 32-key blocks.
__global__ __launch_bounds__(256) void attn_kernel(const __bf16* __restrict__ Q,
                                                   const __bf16* __restrict__ Kx,
                                                   const __bf16* __restrict__ Vt,
                                                   float* __restrict__ O) {
  const int S = 1024, E = 1024, H = 16, D = 64;
  __shared__ __bf16 plds[8][16 * 32];  // per-wave P tile (16x32 bf16)

  const int wv = threadIdx.x >> 5;
  const int lane = threadIdx.x & 31;
  const int hf = lane >> 4;
  const int li = lane & 15;

  const int gw = blockIdx.x * 8 + wv;
  const int qblk = gw & 63;           // S/16 = 64 query blocks
  const int bh = gw >> 6;
  const int b = bh >> 4;
  const int h = bh & 15;
  const int t0 = qblk * 16;

  // Q A-fragments: 16x64 = two 16x32 steps over d
  Frag qa[2];
  {
    const __bf16* qrow = Q + (size_t)(b * S + t0 + li) * E + h * D;
#pragma unroll
    for (int j = 0; j < 2; ++j) {
      qa[j].h8[0] = *(const v8bf*)(qrow + j * 32 + hf * 16);
      qa[j].h8[1] = *(const v8bf*)(qrow + j * 32 + hf * 16 + 8);
    }
  }

  v8f o[4] = {v8f{}, v8f{}, v8f{}, v8f{}};
  float mrow[8], lrow[8];
#pragma unroll
  for (int r = 0; r < 8; ++r) { mrow[r] = -3.0e38f; lrow[r] = 0.0f; }

  const __bf16* vbase = Vt + ((size_t)(b * H + h) * D) * S;
  __bf16* pw = &plds[wv][0];

  const int nkb = (t0 + 47) >> 5;  // 32-key blocks covering keys [0, t0+15]
  for (int kb = 0; kb < nkb; ++kb) {
    const int s0 = kb * 32;

    // ---- scores: two 16x16 tiles (key sub-blocks), K-dim = d (2 steps) ----
    v8f sc[2];
#pragma unroll
    for (int sb = 0; sb < 2; ++sb) {
      const __bf16* krow = Kx + (size_t)(b * S + s0 + sb * 16 + li) * E + h * D;
      v8f s = {};
#pragma unroll
      for (int j = 0; j < 2; ++j) {
        Frag kf;  // B layout: col = key li, K in {j*32 + hf*16 .. +16} over d
        kf.h8[0] = *(const v8bf*)(krow + j * 32 + hf * 16);
        kf.h8[1] = *(const v8bf*)(krow + j * 32 + hf * 16 + 8);
        s = __builtin_amdgcn_wmma_f32_16x16x32_bf16(
            false, qa[j].v, false, kf.v, (short)0, s, false, false);
      }
      sc[sb] = s;
    }

    // ---- causal mask + online softmax (row stats per C-tile VGPR row) ----
    const int tBase = t0 + hf * 8;   // + r = query row of this lane's VGPR r
    const int c0 = s0 + li;
    const int c1 = s0 + 16 + li;
    float p0[8], p1[8], alpha[8];
#pragma unroll
    for (int r = 0; r < 8; ++r) {
      float x0 = (c0 > tBase + r) ? -3.0e38f : sc[0][r];
      float x1 = (c1 > tBase + r) ? -3.0e38f : sc[1][r];
      float mx = fmaxf(x0, x1);
#pragma unroll
      for (int msk = 8; msk; msk >>= 1) mx = fmaxf(mx, __shfl_xor(mx, msk, 32));
      float mnew = fmaxf(mrow[r], mx);
      float a = __expf(mrow[r] - mnew);
      float e0 = __expf(x0 - mnew);
      float e1 = __expf(x1 - mnew);
      float rs = e0 + e1;
#pragma unroll
      for (int msk = 8; msk; msk >>= 1) rs += __shfl_xor(rs, msk, 32);
      lrow[r] = lrow[r] * a + rs;
      mrow[r] = mnew;
      alpha[r] = a; p0[r] = e0; p1[r] = e1;
    }
#pragma unroll
    for (int dt = 0; dt < 4; ++dt)
#pragma unroll
      for (int r = 0; r < 8; ++r) o[dt][r] *= alpha[r];

    // ---- C-layout P -> A-fragment via wave-private LDS (16x32 bf16) ----
#pragma unroll
    for (int r = 0; r < 8; ++r) {
      pw[(hf * 8 + r) * 32 + li]      = f2bf(p0[r]);
      pw[(hf * 8 + r) * 32 + 16 + li] = f2bf(p1[r]);
    }
    Frag pa;  // A layout: row = li, K in {hf*16..+16} over s
    pa.h8[0] = *(const v8bf*)&pw[li * 32 + hf * 16];
    pa.h8[1] = *(const v8bf*)&pw[li * 32 + hf * 16 + 8];

    // ---- O += P x V  (B-frag cols are d, K = s contiguous in Vt) ----
#pragma unroll
    for (int dt = 0; dt < 4; ++dt) {
      const __bf16* vcol = vbase + (size_t)(dt * 16 + li) * S + s0 + hf * 16;
      Frag vf;
      vf.h8[0] = *(const v8bf*)(vcol);
      vf.h8[1] = *(const v8bf*)(vcol + 8);
      o[dt] = __builtin_amdgcn_wmma_f32_16x16x32_bf16(
          false, pa.v, false, vf.v, (short)0, o[dt], false, false);
    }
  }

  // ---- normalize and store fp32 O[b, t, h*64 + d] ----
#pragma unroll
  for (int r = 0; r < 8; ++r) {
    float inv = 1.0f / lrow[r];
    float* orow = O + (size_t)(b * S + t0 + hf * 8 + r) * E + h * D + li;
    orow[0]  = o[0][r] * inv;
    orow[16] = o[1][r] * inv;
    orow[32] = o[2][r] * inv;
    orow[48] = o[3][r] * inv;
  }
}

// --------------------------- LayerNorm (row of 1024) ------------------------
__global__ __launch_bounds__(256) void ln_kernel(const float* __restrict__ X,
                                                 const float* __restrict__ gamma,
                                                 const float* __restrict__ beta,
                                                 __bf16* __restrict__ Y) {
  const int E = 1024;
  const int row = blockIdx.x;
  const int tid = threadIdx.x;
  __shared__ float ssum[8], ssq[8], bc[2];

  float4 xv = *(const float4*)(X + (size_t)row * E + tid * 4);
  float s  = xv.x + xv.y + xv.z + xv.w;
  float sq = xv.x * xv.x + xv.y * xv.y + xv.z * xv.z + xv.w * xv.w;
#pragma unroll
  for (int m = 16; m; m >>= 1) { s += __shfl_xor(s, m, 32); sq += __shfl_xor(sq, m, 32); }
  if ((tid & 31) == 0) { ssum[tid >> 5] = s; ssq[tid >> 5] = sq; }
  __syncthreads();
  if (tid == 0) {
    float ts = 0.f, tq = 0.f;
    for (int i = 0; i < 8; ++i) { ts += ssum[i]; tq += ssq[i]; }
    float mu = ts * (1.0f / E);
    float var = tq * (1.0f / E) - mu * mu;
    bc[0] = mu; bc[1] = rsqrtf(var + 1e-5f);
  }
  __syncthreads();
  float mu = bc[0], rs = bc[1];
  const float* g = gamma + tid * 4;
  const float* bb = beta + tid * 4;
  v4bf y;
  y.x = f2bf((xv.x - mu) * rs * g[0] + bb[0]);
  y.y = f2bf((xv.y - mu) * rs * g[1] + bb[1]);
  y.z = f2bf((xv.z - mu) * rs * g[2] + bb[2]);
  y.w = f2bf((xv.w - mu) * rs * g[3] + bb[3]);
  ((v4bf*)(Y + (size_t)row * E))[tid] = y;
}

// ---------------------------------------------------------------------------
extern "C" void kernel_launch(void* const* d_in, const int* in_sizes, int n_in,
                              void* d_out, int out_size, void* d_ws, size_t ws_size,
                              hipStream_t stream) {
  const float* x     = (const float*)d_in[0];
  const float* Wq    = (const float*)d_in[1];
  const float* Wk    = (const float*)d_in[2];
  const float* Wv    = (const float*)d_in[3];
  const float* Wo    = (const float*)d_in[4];
  const float* gamma = (const float*)d_in[5];
  const float* beta  = (const float*)d_in[6];
  float* out = (float*)d_out;

  const int Mrows = 4096;   // B*S
  const int E = 1024;

  char* ws = (char*)d_ws;
  size_t off = 0;
  auto take = [&](size_t bytes) -> char* {
    char* p = ws + off;
    off += (bytes + 255) & ~(size_t)255;
    return p;
  };
  __bf16* xb  = (__bf16*)take((size_t)Mrows * E * 2);
  __bf16* wqb = (__bf16*)take((size_t)E * E * 2);
  __bf16* wkb = (__bf16*)take((size_t)E * E * 2);
  __bf16* wvb = (__bf16*)take((size_t)E * E * 2);
  __bf16* wob = (__bf16*)take((size_t)E * E * 2);
  __bf16* qb  = (__bf16*)take((size_t)Mrows * E * 2);
  __bf16* kb  = (__bf16*)take((size_t)Mrows * E * 2);
  __bf16* vtb = (__bf16*)take((size_t)Mrows * E * 2);
  float*  of  = (float*)take((size_t)Mrows * E * 4);
  __bf16* lnb = (__bf16*)take((size_t)Mrows * E * 2);

  // fp32 -> bf16 converts
  {
    int n4 = Mrows * E / 4;
    cvt_bf16<<<n4 / 256, 256, 0, stream>>>(x, xb, n4);
    int w4 = E * E / 4;
    cvt_bf16<<<w4 / 256, 256, 0, stream>>>(Wq, wqb, w4);
    cvt_bf16<<<w4 / 256, 256, 0, stream>>>(Wk, wkb, w4);
    cvt_bf16<<<w4 / 256, 256, 0, stream>>>(Wv, wvb, w4);
    cvt_bf16<<<w4 / 256, 256, 0, stream>>>(Wo, wob, w4);
  }

  // projections: blocks = (M/128)*(N/128) = 32*8 = 256, 256 threads each
  const int gemmBlocks = (Mrows / 128) * (E / 128);
  gemm_xwT<1><<<gemmBlocks, 256, 0, stream>>>(xb, wqb, qb, Mrows, E, E, 1.0f);
  gemm_xwT<1><<<gemmBlocks, 256, 0, stream>>>(xb, wkb, kb, Mrows, E, E, 0.125f);  // 1/sqrt(64)
  gemm_xwT<2><<<gemmBlocks, 256, 0, stream>>>(xb, wvb, vtb, Mrows, E, E, 1.0f);

  // attention: B*H*(S/16) = 4096 waves -> 512 blocks
  attn_kernel<<<512, 256, 0, stream>>>(qb, kb, vtb, of);

  // layernorm: one block per row
  ln_kernel<<<Mrows, 256, 0, stream>>>(of, gamma, beta, lnb);

  // output projection -> fp32 d_out
  gemm_xwT<0><<<gemmBlocks, 256, 0, stream>>>(lnb, wob, out, Mrows, E, E, 1.0f);
}
